// SceneGraphGNN_1468878815658
// MI455X (gfx1250) — matrix operand order repaired and checked by reading, not compiled, not run on hardware
//
#include <hip/hip_runtime.h>
#include <hip/hip_bf16.h>
#include <math.h>

// SceneGraph GNN for MI455X (gfx1250, wave32).
// Bandwidth-bound (~2.6 GB of edge-gather/scatter traffic -> ~120us @ 23.3TB/s).
// All K>=64 GEMMs use v_wmma_f32_16x16x32_f16 (f16 in / f32 accumulate),
// weights staged transposed in LDS, segment-sum via global_atomic_add_f32
// directly from WMMA C fragments.

typedef __attribute__((ext_vector_type(16))) _Float16 v16h;
typedef __attribute__((ext_vector_type(2)))  _Float16 h2;
typedef __attribute__((ext_vector_type(8)))  float    v8f;

#define NNODES 100000
#define NEDGES 1000000

// ---------- WMMA fragment helpers (CDNA5 16-bit A 16x32 layout) ----------
// A: lane l holds row m=l&15; VGPR r (pairs): K = kbase + (r<4 ? 2r+8*hi : 16+2(r-4)+8*hi)
static __device__ __forceinline__ int a_kidx(int kbase, int r, int hi) {
  return kbase + ((r < 4) ? (2 * r + 8 * hi) : (16 + 2 * (r - 4) + 8 * hi));
}

static __device__ __forceinline__ v16h load_a_concat64(const _Float16* rowA,
                                                       const _Float16* rowB,
                                                       int kbase, int hi) {
  v16h a;
#pragma unroll
  for (int r = 0; r < 8; ++r) {
    int k = a_kidx(kbase, r, hi);                 // always even; pair never straddles 64
    h2 p = (k < 64) ? *(const h2*)(rowA + k) : *(const h2*)(rowB + (k - 64));
    a[2 * r] = p[0]; a[2 * r + 1] = p[1];
  }
  return a;
}

static __device__ __forceinline__ v16h load_a_row(const _Float16* row, int kbase, int hi) {
  v16h a;
#pragma unroll
  for (int r = 0; r < 8; ++r) {
    int k = a_kidx(kbase, r, hi);
    h2 p = *(const h2*)(row + k);
    a[2 * r] = p[0]; a[2 * r + 1] = p[1];
  }
  return a;
}

// B 32x16: lane l holds column n=l&15; lanes 0-15 K=kbase+{0..7,16..23}, lanes 16-31 +8.
// LDS image is W transposed: Wl[n*ldsk + k]  (ldsk = Kpad+8 to break bank conflicts)
static __device__ __forceinline__ v16h load_b_lds(const _Float16* Wl, int ldsk,
                                                  int n, int kbase, int hi) {
  v16h b;
  const _Float16* col = Wl + n * ldsk + kbase + hi * 8;
#pragma unroll
  for (int v = 0; v < 8; ++v) {
    h2 p = *(const h2*)(col + 2 * v + (v < 4 ? 0 : 8));
    b[2 * v] = p[0]; b[2 * v + 1] = p[1];
  }
  return b;
}

static __device__ __forceinline__ float wave_allsum(float v) {
#pragma unroll
  for (int m = 16; m >= 1; m >>= 1) v += __shfl_xor(v, m, 32);
  return v;
}

// ---------- weight prep: WT[n][kp] = (kp<K ? W[kp][n] : 0), f32 -> f16 ----------
__global__ void k_convT(const float* __restrict__ W, _Float16* __restrict__ WT,
                        int K, int Nn, int Kpad) {
  int i = blockIdx.x * blockDim.x + threadIdx.x;
  if (i >= Nn * Kpad) return;
  int n = i / Kpad, kp = i % Kpad;
  WT[i] = (kp < K) ? (_Float16)W[kp * Nn + n] : (_Float16)0.0f;
}

// ---------- encoders: LN(relu(X @ W + b)); one wave per row, 2 cols/lane ----------
template <int IN_DIM>
__global__ void __launch_bounds__(256) k_encode(
    const float* __restrict__ X, const float* __restrict__ W,
    const float* __restrict__ B, const float* __restrict__ G,
    const float* __restrict__ Beta, float* __restrict__ out32,
    _Float16* __restrict__ out16, int nRows) {
  int wave = threadIdx.x >> 5, lane = threadIdx.x & 31;
  int row = blockIdx.x * 8 + wave;
  if (row >= nRows) return;
  float xin[IN_DIM];
#pragma unroll
  for (int d = 0; d < IN_DIM; ++d) xin[d] = X[(size_t)row * IN_DIM + d];
  int j0 = lane, j1 = lane + 32;
  float a0 = B[j0], a1 = B[j1];
#pragma unroll
  for (int d = 0; d < IN_DIM; ++d) {
    a0 += xin[d] * W[d * 64 + j0];
    a1 += xin[d] * W[d * 64 + j1];
  }
  a0 = fmaxf(a0, 0.0f); a1 = fmaxf(a1, 0.0f);
  float mu  = wave_allsum(a0 + a1) * (1.0f / 64.0f);
  float d0 = a0 - mu, d1 = a1 - mu;
  float var = wave_allsum(d0 * d0 + d1 * d1) * (1.0f / 64.0f);
  float inv = rsqrtf(var + 1e-5f);
  float o0 = d0 * inv * G[j0] + Beta[j0];
  float o1 = d1 * inv * G[j1] + Beta[j1];
  if (out32) {
    out32[(size_t)row * 64 + j0] = o0;
    out32[(size_t)row * 64 + j1] = o1;
  }
  out16[(size_t)row * 64 + j0] = (_Float16)o0;
  out16[(size_t)row * 64 + j1] = (_Float16)o1;
}

__global__ void k_zero(float* p, size_t n) {
  size_t i = (size_t)blockIdx.x * blockDim.x + threadIdx.x;
  if (i < n) p[i] = 0.0f;
}

__global__ void k_count(const int* __restrict__ tgt, float* __restrict__ cnt, int nE) {
  int i = blockIdx.x * blockDim.x + threadIdx.x;
  if (i < nE) atomicAdd(&cnt[tgt[i]], 1.0f);
}

// ---------- msg GEMM: relu([h[src]|e_enc] @ Wmsg + b) scattered into msum[tgt] ----------
__global__ void __launch_bounds__(256) k_msg_gemm(
    const _Float16* __restrict__ hF, const _Float16* __restrict__ eF,
    const int* __restrict__ src, const int* __restrict__ tgt,
    const _Float16* __restrict__ WT, const float* __restrict__ bias,
    float* __restrict__ msum, int nE) {
  const int KP = 128, LDSK = KP + 8;
  __shared__ _Float16 Wl[64 * (128 + 8)];
  for (int i = threadIdx.x; i < 64 * KP / 2; i += 256) {
    int n = i / (KP / 2), k2 = i % (KP / 2);
    *(h2*)(Wl + n * LDSK + 2 * k2) = *(const h2*)(WT + n * KP + 2 * k2);
  }
  __syncthreads();
  int wave = threadIdx.x >> 5, lane = threadIdx.x & 31;
  int tile = blockIdx.x * 8 + wave;
  if (tile * 16 >= nE) return;
  int e0 = tile * 16;
  int m = lane & 15, hi = lane >> 4;
  int e = e0 + m;
  const _Float16* rowH = hF + (size_t)src[e] * 64;
  const _Float16* rowE = eF + (size_t)e * 64;
  v16h A[4];
#pragma unroll
  for (int kt = 0; kt < 4; ++kt) A[kt] = load_a_concat64(rowH, rowE, kt * 32, hi);
  int tg[8];
#pragma unroll
  for (int r = 0; r < 8; ++r) tg[r] = tgt[e0 + r + 8 * hi];
#pragma unroll
  for (int nt = 0; nt < 4; ++nt) {
    int n = m + nt * 16;
    float bv = bias[n];
    v8f c;
#pragma unroll
    for (int i = 0; i < 8; ++i) c[i] = bv;
#pragma unroll
    for (int kt = 0; kt < 4; ++kt) {
      v16h b = load_b_lds(Wl, LDSK, n, kt * 32, hi);
      c = __builtin_amdgcn_wmma_f32_16x16x32_f16(false, A[kt], false, b,
                                                 (short)0, c, false, false);
    }
#pragma unroll
    for (int r = 0; r < 8; ++r) {          // C layout: row = r + 8*hi, col = n
      float v = fmaxf(c[r], 0.0f);
      atomicAdd(&msum[(size_t)tg[r] * 64 + n], v);
    }
  }
}

__global__ void k_mean(const float* __restrict__ msum, const float* __restrict__ cnt,
                       _Float16* __restrict__ mean16, int nN) {
  int i = blockIdx.x * blockDim.x + threadIdx.x;
  if (i < nN * 64) mean16[i] = (_Float16)(msum[i] / fmaxf(cnt[i >> 6], 1.0f));
}

// ---------- update GEMM: u = [h|mean] @ Wupd + b (pre-LN, f32 out) ----------
__global__ void __launch_bounds__(256) k_upd_gemm(
    const _Float16* __restrict__ hF, const _Float16* __restrict__ meanF,
    const _Float16* __restrict__ WT, const float* __restrict__ bias,
    float* __restrict__ u, int nN) {
  const int KP = 128, LDSK = KP + 8;
  __shared__ _Float16 Wl[64 * (128 + 8)];
  for (int i = threadIdx.x; i < 64 * KP / 2; i += 256) {
    int n = i / (KP / 2), k2 = i % (KP / 2);
    *(h2*)(Wl + n * LDSK + 2 * k2) = *(const h2*)(WT + n * KP + 2 * k2);
  }
  __syncthreads();
  int wave = threadIdx.x >> 5, lane = threadIdx.x & 31;
  int tile = blockIdx.x * 8 + wave;
  if (tile * 16 >= nN) return;
  int r0 = tile * 16;
  int m = lane & 15, hi = lane >> 4;
  int node = r0 + m;
  const _Float16* rowH = hF + (size_t)node * 64;
  const _Float16* rowM = meanF + (size_t)node * 64;
  v16h A[4];
#pragma unroll
  for (int kt = 0; kt < 4; ++kt) A[kt] = load_a_concat64(rowH, rowM, kt * 32, hi);
#pragma unroll
  for (int nt = 0; nt < 4; ++nt) {
    int n = m + nt * 16;
    float bv = bias[n];
    v8f c;
#pragma unroll
    for (int i = 0; i < 8; ++i) c[i] = bv;
#pragma unroll
    for (int kt = 0; kt < 4; ++kt) {
      v16h b = load_b_lds(Wl, LDSK, n, kt * 32, hi);
      c = __builtin_amdgcn_wmma_f32_16x16x32_f16(false, A[kt], false, b,
                                                 (short)0, c, false, false);
    }
#pragma unroll
    for (int r = 0; r < 8; ++r)
      u[(size_t)(r0 + r + 8 * hi) * 64 + n] = c[r];
  }
}

// ---------- h = relu(LN(u) + h); one wave per node ----------
__global__ void __launch_bounds__(256) k_ln_res(
    const float* __restrict__ u, const float* __restrict__ G,
    const float* __restrict__ Beta, float* __restrict__ h32,
    _Float16* __restrict__ h16, int nN) {
  int wave = threadIdx.x >> 5, lane = threadIdx.x & 31;
  int row = blockIdx.x * 8 + wave;
  if (row >= nN) return;
  size_t base = (size_t)row * 64;
  float a0 = u[base + lane], a1 = u[base + lane + 32];
  float mu  = wave_allsum(a0 + a1) * (1.0f / 64.0f);
  float d0 = a0 - mu, d1 = a1 - mu;
  float var = wave_allsum(d0 * d0 + d1 * d1) * (1.0f / 64.0f);
  float inv = rsqrtf(var + 1e-5f);
  float v0 = d0 * inv * G[lane] + Beta[lane];
  float v1 = d1 * inv * G[lane + 32] + Beta[lane + 32];
  float n0 = fmaxf(v0 + h32[base + lane], 0.0f);
  float n1 = fmaxf(v1 + h32[base + lane + 32], 0.0f);
  h32[base + lane] = n0;        h32[base + lane + 32] = n1;
  h16[base + lane] = (_Float16)n0; h16[base + lane + 32] = (_Float16)n1;
}

// ---------- predictor stage 1: z1 = relu([h[src]|h[tgt]|attr|0pad]@Wp1+b), K=160 ----------
static __device__ __forceinline__ _Float16 a_pred(const _Float16* hs, const _Float16* ht,
                                                  const float* at, int k) {
  if (k < 64)  return hs[k];
  if (k < 128) return ht[k - 64];
  if (k < 133) return (_Float16)at[k - 128];
  return (_Float16)0.0f;
}

__global__ void __launch_bounds__(256) k_pred1(
    const _Float16* __restrict__ hF, const int* __restrict__ src,
    const int* __restrict__ tgt, const float* __restrict__ attr,
    const _Float16* __restrict__ WT, const float* __restrict__ bias,
    _Float16* __restrict__ z1, int nE) {
  const int KP = 160, LDSK = 168;
  __shared__ _Float16 Wl[64 * 168];
  for (int i = threadIdx.x; i < 64 * KP / 2; i += 256) {
    int n = i / (KP / 2), k2 = i % (KP / 2);
    *(h2*)(Wl + n * LDSK + 2 * k2) = *(const h2*)(WT + n * KP + 2 * k2);
  }
  __syncthreads();
  int wave = threadIdx.x >> 5, lane = threadIdx.x & 31;
  int tile = blockIdx.x * 8 + wave;
  if (tile * 16 >= nE) return;
  int e0 = tile * 16;
  int m = lane & 15, hi = lane >> 4;
  int e = e0 + m;
  const _Float16* hs = hF + (size_t)src[e] * 64;
  const _Float16* ht = hF + (size_t)tgt[e] * 64;
  const float* at = attr + (size_t)e * 5;
  v16h A[5];
#pragma unroll
  for (int kt = 0; kt < 5; ++kt) {
#pragma unroll
    for (int r = 0; r < 8; ++r) {
      int k = a_kidx(kt * 32, r, hi);
      A[kt][2 * r]     = a_pred(hs, ht, at, k);
      A[kt][2 * r + 1] = a_pred(hs, ht, at, k + 1);
    }
  }
#pragma unroll
  for (int nt = 0; nt < 4; ++nt) {
    int n = m + nt * 16;
    float bv = bias[n];
    v8f c;
#pragma unroll
    for (int i = 0; i < 8; ++i) c[i] = bv;
#pragma unroll
    for (int kt = 0; kt < 5; ++kt) {
      v16h b = load_b_lds(Wl, LDSK, n, kt * 32, hi);
      c = __builtin_amdgcn_wmma_f32_16x16x32_f16(false, A[kt], false, b,
                                                 (short)0, c, false, false);
    }
#pragma unroll
    for (int r = 0; r < 8; ++r)
      z1[(size_t)(e0 + r + 8 * hi) * 64 + n] = (_Float16)fmaxf(c[r], 0.0f);
  }
}

// ---------- predictor stage 2: z2 = relu(z1 @ Wp2 + b), K=64, Nout=32 ----------
__global__ void __launch_bounds__(256) k_pred2(
    const _Float16* __restrict__ z1, const _Float16* __restrict__ WT,
    const float* __restrict__ bias, _Float16* __restrict__ z2, int nE) {
  const int KP = 64, LDSK = 72;
  __shared__ _Float16 Wl[32 * 72];
  for (int i = threadIdx.x; i < 32 * KP / 2; i += 256) {
    int n = i / (KP / 2), k2 = i % (KP / 2);
    *(h2*)(Wl + n * LDSK + 2 * k2) = *(const h2*)(WT + n * KP + 2 * k2);
  }
  __syncthreads();
  int wave = threadIdx.x >> 5, lane = threadIdx.x & 31;
  int tile = blockIdx.x * 8 + wave;
  if (tile * 16 >= nE) return;
  int e0 = tile * 16;
  int m = lane & 15, hi = lane >> 4;
  const _Float16* rowA = z1 + (size_t)(e0 + m) * 64;
  v16h A[2];
#pragma unroll
  for (int kt = 0; kt < 2; ++kt) A[kt] = load_a_row(rowA, kt * 32, hi);
#pragma unroll
  for (int nt = 0; nt < 2; ++nt) {
    int n = m + nt * 16;
    float bv = bias[n];
    v8f c;
#pragma unroll
    for (int i = 0; i < 8; ++i) c[i] = bv;
#pragma unroll
    for (int kt = 0; kt < 2; ++kt) {
      v16h b = load_b_lds(Wl, LDSK, n, kt * 32, hi);
      c = __builtin_amdgcn_wmma_f32_16x16x32_f16(false, A[kt], false, b,
                                                 (short)0, c, false, false);
    }
#pragma unroll
    for (int r = 0; r < 8; ++r)
      z2[(size_t)(e0 + r + 8 * hi) * 32 + n] = (_Float16)fmaxf(c[r], 0.0f);
  }
}

// ---------- predictor stage 3: out = sigmoid(z2 . w_p3 + b_p3) ----------
__global__ void k_pred3(const _Float16* __restrict__ z2, const float* __restrict__ w3,
                        const float* __restrict__ b3, float* __restrict__ out, int nE) {
  int i = blockIdx.x * blockDim.x + threadIdx.x;
  if (i >= nE) return;
  float acc = b3[0];
#pragma unroll
  for (int k = 0; k < 32; ++k) acc += (float)z2[(size_t)i * 32 + k] * w3[k];
  out[i] = 1.0f / (1.0f + expf(-acc));
}

// ==================================================================
extern "C" void kernel_launch(void* const* d_in, const int* in_sizes, int n_in,
                              void* d_out, int out_size, void* d_ws, size_t ws_size,
                              hipStream_t stream) {
  (void)in_sizes; (void)n_in; (void)out_size; (void)ws_size;
  const float* x        = (const float*)d_in[0];
  const int*   ei       = (const int*)d_in[1];
  const float* eattr    = (const float*)d_in[2];
  const float* w_node   = (const float*)d_in[3];
  const float* b_node   = (const float*)d_in[4];
  const float* g_node   = (const float*)d_in[5];
  const float* be_node  = (const float*)d_in[6];
  const float* w_edge   = (const float*)d_in[7];
  const float* b_edge   = (const float*)d_in[8];
  const float* g_edge   = (const float*)d_in[9];
  const float* be_edge  = (const float*)d_in[10];
  const float* w_msg0   = (const float*)d_in[11];
  const float* b_msg0   = (const float*)d_in[12];
  const float* w_upd0   = (const float*)d_in[13];
  const float* b_upd0   = (const float*)d_in[14];
  const float* g0       = (const float*)d_in[15];
  const float* beta0    = (const float*)d_in[16];
  const float* w_msg1   = (const float*)d_in[17];
  const float* b_msg1   = (const float*)d_in[18];
  const float* w_upd1   = (const float*)d_in[19];
  const float* b_upd1   = (const float*)d_in[20];
  const float* g1       = (const float*)d_in[21];
  const float* beta1    = (const float*)d_in[22];
  const float* w_p1     = (const float*)d_in[23];
  const float* b_p1     = (const float*)d_in[24];
  const float* w_p2     = (const float*)d_in[25];
  const float* b_p2     = (const float*)d_in[26];
  const float* w_p3     = (const float*)d_in[27];
  const float* b_p3     = (const float*)d_in[28];
  const int* src = ei;
  const int* tgt = ei + NEDGES;

  char* ws = (char*)d_ws;
  size_t off = 0;
  auto alloc = [&](size_t bytes) -> void* {
    void* p = ws + off;
    off = (off + bytes + 255) & ~((size_t)255);
    return p;
  };
  float*    h32    = (float*)   alloc((size_t)NNODES * 64 * 4);
  _Float16* h16    = (_Float16*)alloc((size_t)NNODES * 64 * 2);
  _Float16* eenc   = (_Float16*)alloc((size_t)NEDGES * 64 * 2);  // reused as z1
  float*    msum   = (float*)   alloc((size_t)NNODES * 64 * 4);
  float*    cnt    = (float*)   alloc((size_t)NNODES * 4);
  _Float16* mean16 = (_Float16*)alloc((size_t)NNODES * 64 * 2);
  float*    ubuf   = (float*)   alloc((size_t)NNODES * 64 * 4);
  _Float16* z2     = (_Float16*)alloc((size_t)NEDGES * 32 * 2);
  _Float16* wmsg0T = (_Float16*)alloc(64 * 128 * 2);
  _Float16* wmsg1T = (_Float16*)alloc(64 * 128 * 2);
  _Float16* wupd0T = (_Float16*)alloc(64 * 128 * 2);
  _Float16* wupd1T = (_Float16*)alloc(64 * 128 * 2);
  _Float16* wp1T   = (_Float16*)alloc(64 * 160 * 2);
  _Float16* wp2T   = (_Float16*)alloc(32 * 64 * 2);
  _Float16* z1     = eenc;

  // weight prep (f32 -> f16, transpose, zero-pad K)
  k_convT<<<(64 * 128 + 255) / 256, 256, 0, stream>>>(w_msg0, wmsg0T, 128, 64, 128);
  k_convT<<<(64 * 128 + 255) / 256, 256, 0, stream>>>(w_msg1, wmsg1T, 128, 64, 128);
  k_convT<<<(64 * 128 + 255) / 256, 256, 0, stream>>>(w_upd0, wupd0T, 128, 64, 128);
  k_convT<<<(64 * 128 + 255) / 256, 256, 0, stream>>>(w_upd1, wupd1T, 128, 64, 128);
  k_convT<<<(64 * 160 + 255) / 256, 256, 0, stream>>>(w_p1,   wp1T,   133, 64, 160);
  k_convT<<<(32 * 64  + 255) / 256, 256, 0, stream>>>(w_p2,   wp2T,   64,  32, 64);

  // encoders
  k_encode<7><<<(NNODES + 7) / 8, 256, 0, stream>>>(x, w_node, b_node, g_node,
                                                    be_node, h32, h16, NNODES);
  k_encode<5><<<(NEDGES + 7) / 8, 256, 0, stream>>>(eattr, w_edge, b_edge, g_edge,
                                                    be_edge, nullptr, eenc, NEDGES);
  // in-degree counts (shared by both conv layers)
  k_zero<<<(NNODES + 255) / 256, 256, 0, stream>>>(cnt, (size_t)NNODES);
  k_count<<<(NEDGES + 255) / 256, 256, 0, stream>>>(tgt, cnt, NEDGES);

  const int eTiles = (NEDGES + 15) / 16, nTiles = (NNODES + 15) / 16;
  const int eBlocks = (eTiles + 7) / 8,  nBlocks = (nTiles + 7) / 8;

  for (int L = 0; L < 2; ++L) {
    const _Float16* wm = L ? wmsg1T : wmsg0T;
    const float*    bm = L ? b_msg1 : b_msg0;
    const _Float16* wu = L ? wupd1T : wupd0T;
    const float*    bu = L ? b_upd1 : b_upd0;
    const float*    g  = L ? g1 : g0;
    const float*    be = L ? beta1 : beta0;
    k_zero<<<(NNODES * 64 + 255) / 256, 256, 0, stream>>>(msum, (size_t)NNODES * 64);
    k_msg_gemm<<<eBlocks, 256, 0, stream>>>(h16, eenc, src, tgt, wm, bm, msum, NEDGES);
    k_mean<<<(NNODES * 64 + 255) / 256, 256, 0, stream>>>(msum, cnt, mean16, NNODES);
    k_upd_gemm<<<nBlocks, 256, 0, stream>>>(h16, mean16, wu, bu, ubuf, NNODES);
    k_ln_res<<<(NNODES + 7) / 8, 256, 0, stream>>>(ubuf, g, be, h32, h16, NNODES);
  }

  // edge predictor
  k_pred1<<<eBlocks, 256, 0, stream>>>(h16, src, tgt, eattr, wp1T, b_p1, z1, NEDGES);
  k_pred2<<<eBlocks, 256, 0, stream>>>(z1, wp2T, b_p2, z2, NEDGES);
  k_pred3<<<(NEDGES + 255) / 256, 256, 0, stream>>>(z2, w_p3, b_p3, (float*)d_out, NEDGES);
}
